// Model_2173253452122
// MI455X (gfx1250) — matrix operand order repaired and checked by reading, not compiled, not run on hardware
//
#include <hip/hip_runtime.h>

typedef __attribute__((ext_vector_type(16))) __bf16 v16bf;
typedef __attribute__((ext_vector_type(8)))  __bf16 v8bf;
typedef __attribute__((ext_vector_type(8)))  float  v8f;

#define IMG_H 256
#define IMG_W 256
#define WIN   21
#define HALFW 10
#define KWIN  441            // real window elements
#define KPAD  448            // padded K/J dim (14 * 32)
#define PIX   128            // pixels per workgroup (half image row)
#define COLS  148            // PIX + 2*HALFW
#define WSTR  456            // bf16 stride per pixel row in LDS (bank-conflict-free)
#define NTHR  256            // 8 wave32
#define BIGF  1e30f

// ---- dynamic LDS carve (bytes) ----
#define OFF_W   0
#define SZ_W    (PIX * WSTR * 2)        // 116736  bf16 w-matrix tile [pix][KPAD(+pad)]
#define OFF_G   (OFF_W + SZ_W)          // 116736
#define SZ_G    (WIN * COLS * 4)        // 12432   f32 guide window
#define OFF_B   (OFF_G + SZ_G)          // 129168
#define SZ_B    (KPAD * 4)              // 1792    f32 breg (padded)
#define OFF_F   (OFF_B + SZ_B)          // 130960
#define SZ_F    (3 * WIN * COLS * 2)    // 18648   bf16 feature window
#define SMEM_SZ (OFF_F + SZ_F)          // 149608  < 160 KB

__device__ __forceinline__ int reflect_idx(int i, int n) {
    i = (i < 0) ? -i : i;
    i = (i >= n) ? (2 * n - 2 - i) : i;
    return i;
}

// ---------------- prep: Wreg f32[441][441] -> bf16[448][448] (zero padded) --------------
__global__ void wreg_to_bf16(const float* __restrict__ Wreg, __bf16* __restrict__ Wb) {
    int idx = blockIdx.x * blockDim.x + threadIdx.x;
    if (idx >= KPAD * KPAD) return;
    int j = idx / KPAD;
    int k = idx - j * KPAD;
    float v = (j < KWIN && k < KWIN) ? Wreg[j * KWIN + k] : 0.0f;
    Wb[idx] = (__bf16)v;
}

// ---------------- main fused NLM kernel ----------------
__launch_bounds__(NTHR)
__global__ void nlm_kernel(const float* __restrict__ ycbcr,
                           const float* __restrict__ feat_y,
                           const float* __restrict__ feat_cb,
                           const float* __restrict__ feat_cr,
                           const __bf16* __restrict__ Wb,
                           const float* __restrict__ breg,
                           float* __restrict__ out) {
    extern __shared__ char smem[];
    __bf16* s_w    = (__bf16*)(smem + OFF_W);
    float*  s_gd   = (float*)(smem + OFF_G);
    float*  s_breg = (float*)(smem + OFF_B);
    __bf16* s_feat = (__bf16*)(smem + OFF_F);

    const int x0 = blockIdx.x * PIX;     // 0 or 128
    const int y  = blockIdx.y;           // image row
    const int ch = blockIdx.z;           // channel
    const int t  = threadIdx.x;

    const float* feat  = (ch == 0) ? feat_y : (ch == 1) ? feat_cb : feat_cr;
    const float* guide = ycbcr + ch * (IMG_H * IMG_W);

    // ---- phase 1: stage reflected windows into LDS ----
    for (int idx = t; idx < 3 * WIN * COLS; idx += NTHR) {
        int c   = idx / (WIN * COLS);
        int rem = idx - c * (WIN * COLS);
        int r   = rem / COLS;
        int col = rem - r * COLS;
        int gy = reflect_idx(y - HALFW + r, IMG_H);
        int gx = reflect_idx(x0 - HALFW + col, IMG_W);
        s_feat[idx] = (__bf16)feat[(c * IMG_H + gy) * IMG_W + gx];
    }
    for (int idx = t; idx < WIN * COLS; idx += NTHR) {
        int r   = idx / COLS;
        int col = idx - r * COLS;
        int gy = reflect_idx(y - HALFW + r, IMG_H);
        int gx = reflect_idx(x0 - HALFW + col, IMG_W);
        s_gd[idx] = guide[gy * IMG_W + gx];
    }
    for (int idx = t; idx < KPAD; idx += NTHR)
        s_breg[idx] = (idx < KWIN) ? breg[idx] : 0.0f;
    __syncthreads();

    // ---- phase 2: w[k,p] = mean_c (feat[c,win_k] - feat[c,center])^2  -> bf16 LDS ----
    {
        const int p     = t & (PIX - 1);
        const int kbase = (t >> 7) * 224;          // [0,224) or [224,448)
        const float c0 = (float)s_feat[(0 * WIN + HALFW) * COLS + p + HALFW];
        const float c1 = (float)s_feat[(1 * WIN + HALFW) * COLS + p + HALFW];
        const float c2 = (float)s_feat[(2 * WIN + HALFW) * COLS + p + HALFW];
        for (int kk = 0; kk < 224; kk += 2) {
            float wv[2];
#pragma unroll
            for (int u = 0; u < 2; ++u) {
                int k = kbase + kk + u;
                float w = 0.0f;
                if (k < KWIN) {
                    int ry = k / WIN;
                    int dx = k - ry * WIN;
                    int cx = p + dx;               // local window column
                    float d0 = (float)s_feat[(0 * WIN + ry) * COLS + cx] - c0;
                    float d1 = (float)s_feat[(1 * WIN + ry) * COLS + cx] - c1;
                    float d2 = (float)s_feat[(2 * WIN + ry) * COLS + cx] - c2;
                    w = (d0 * d0 + d1 * d1 + d2 * d2) * (1.0f / 3.0f);
                }
                wv[u] = w;
            }
            union { __bf16 h[2]; unsigned u32; } pk;
            pk.h[0] = (__bf16)wv[0];
            pk.h[1] = (__bf16)wv[1];
            *(unsigned*)(&s_w[p * WSTR + kbase + kk]) = pk.u32;
        }
    }
    __syncthreads();

    // ---- phase 3: per-wave 16-pixel N-tile, GEMM via WMMA + fused online softmax ----
    const int wave = t >> 5;
    const int lane = t & 31;
    const int ln   = lane & 15;
    const int hi   = lane >> 4;          // 0: lanes 0-15, 1: lanes 16-31
    const int p0   = wave * 16;          // local pixel base of this wave's N-tile
    const int myp  = p0 + ln;            // local pixel this lane tracks

    // cache all 14 B fragments (w-matrix, K x N = 32 x 16 per step) in registers.
    // B layout: lane L elem e -> N = L&15, K = 32*kt + 16*(L>>4) + e  (32 contiguous halves)
    v16bf bfrag[14];
    {
        const __bf16* wrow = &s_w[myp * WSTR];
#pragma unroll
        for (int kt = 0; kt < 14; ++kt) {
            const __bf16* ptr = wrow + kt * 32 + hi * 16;
            v8bf lo = *(const v8bf*)(ptr);
            v8bf hq = *(const v8bf*)(ptr + 8);
            bfrag[kt] = __builtin_shufflevector(lo, hq, 0, 1, 2, 3, 4, 5, 6, 7,
                                                8, 9, 10, 11, 12, 13, 14, 15);
        }
    }

    float m = -BIGF, ssum = 0.0f, osum = 0.0f;   // online softmax state (per pixel col)

    for (int jt = 0; jt < 28; ++jt) {
        v8f acc0 = {0.f, 0.f, 0.f, 0.f, 0.f, 0.f, 0.f, 0.f};
        v8f acc1 = {0.f, 0.f, 0.f, 0.f, 0.f, 0.f, 0.f, 0.f};
        // A layout: lane L elem e -> M = L&15, K = 32*kt + 16*(L>>4) + e
        const __bf16* arow = Wb + (jt * 16 + ln) * KPAD;
#pragma unroll
        for (int kt = 0; kt < 14; ++kt) {
            const __bf16* pa = arow + kt * 32 + hi * 16;
            v8bf alo = *(const v8bf*)(pa);        // K chunk [base, base+8)
            v8bf ahi = *(const v8bf*)(pa + 8);    // K chunk [base+8, base+16)
            v16bf afrag = __builtin_shufflevector(alo, ahi, 0, 1, 2, 3, 4, 5, 6, 7,
                                                  8, 9, 10, 11, 12, 13, 14, 15);
            if (kt & 1)
                acc1 = __builtin_amdgcn_wmma_f32_16x16x32_bf16(
                    false, afrag, false, bfrag[kt], (short)0, acc1, false, false);
            else
                acc0 = __builtin_amdgcn_wmma_f32_16x16x32_bf16(
                    false, afrag, false, bfrag[kt], (short)0, acc0, false, false);
        }

        // C layout: lane L vgpr r -> j = jt*16 + r + 8*(L>>4), pixel = p0 + (L&15)
        float logit[8];
        float tmax = -BIGF;
#pragma unroll
        for (int r = 0; r < 8; ++r) {
            int j = jt * 16 + r + 8 * hi;
            float a = acc0[r] + acc1[r];
            float l = (j < KWIN) ? -(a + s_breg[j]) : -BIGF;
            logit[r] = l;
            tmax = fmaxf(tmax, l);
        }
        tmax = fmaxf(tmax, __shfl_xor(tmax, 16, 32));
        float newm  = fmaxf(m, tmax);
        float alpha = __expf(m - newm);
        float ls = 0.0f, lo = 0.0f;
#pragma unroll
        for (int r = 0; r < 8; ++r) {
            int j  = jt * 16 + r + 8 * hi;
            int jc = (j < KWIN) ? j : (KWIN - 1);
            float e  = __expf(logit[r] - newm);   // exactly 0 for masked j
            int ry = jc / WIN;
            int dx = jc - ry * WIN;
            float g = s_gd[ry * COLS + myp + dx]; // guide patch element y_p(j)
            ls += e;
            lo += e * g;
        }
        ls += __shfl_xor(ls, 16, 32);
        lo += __shfl_xor(lo, 16, 32);
        ssum = ssum * alpha + ls;
        osum = osum * alpha + lo;
        m = newm;
    }

    if (hi == 0) {
        int gx = x0 + myp;
        out[(ch * IMG_H + y) * IMG_W + gx] = osum / ssum;
    }
}

extern "C" void kernel_launch(void* const* d_in, const int* in_sizes, int n_in,
                              void* d_out, int out_size, void* d_ws, size_t ws_size,
                              hipStream_t stream) {
    (void)in_sizes; (void)n_in; (void)out_size; (void)ws_size;
    const float* ycbcr   = (const float*)d_in[0];
    const float* feat_y  = (const float*)d_in[1];
    const float* feat_cb = (const float*)d_in[2];
    const float* feat_cr = (const float*)d_in[3];
    const float* Wreg    = (const float*)d_in[4];
    const float* breg    = (const float*)d_in[5];
    __bf16* Wb = (__bf16*)d_ws;                 // 448*448*2 = 401,408 B of scratch
    float* out = (float*)d_out;

    wreg_to_bf16<<<(KPAD * KPAD + 255) / 256, 256, 0, stream>>>(Wreg, Wb);

    dim3 grid(IMG_W / PIX, IMG_H, 3);
    nlm_kernel<<<grid, NTHR, SMEM_SZ, stream>>>(ycbcr, feat_y, feat_cb, feat_cr,
                                                Wb, breg, out);
}
